// GIN_45397804319031
// MI455X (gfx1250) — matrix operand order repaired and checked by reading, not compiled, not run on hardware
//
#include <hip/hip_runtime.h>

#define NN   10000     // nodes
#define NE   640000    // edges
#define DD   128       // feature dim (D == H)
#define HID2 256       // MLP hidden (2*H)
#define NG   200       // graphs
#define NO   10        // classes

typedef __attribute__((ext_vector_type(16))) __bf16 v16bf;
typedef __attribute__((ext_vector_type(8)))  float  v8f;

union FragAB { v16bf v; uint4 q[2]; };
union FragC  { v8f   v; float f[8]; };

__device__ __forceinline__ unsigned short f2bf(float x) {
  unsigned int u = __float_as_uint(x);
  u += 0x7FFFu + ((u >> 16) & 1u);            // round-to-nearest-even
  return (unsigned short)(u >> 16);
}

// ---------------------------------------------------------------- utilities
__global__ void zero_f32(float* __restrict__ p, int n) {
  int i = blockIdx.x * blockDim.x + threadIdx.x;
  if (i < n) p[i] = 0.0f;
}

// one wave per edge; each lane handles 4 of the 128 features (coalesced)
__global__ void scatter_add(const float* __restrict__ x,
                            const int* __restrict__ src,
                            const int* __restrict__ dst,
                            float* __restrict__ agg) {
  int gid  = blockIdx.x * blockDim.x + threadIdx.x;
  int e    = gid >> 5;
  int lane = gid & 31;
  if (e >= NE) return;
  int s = src[e] * DD;
  int d = dst[e] * DD;
#pragma unroll
  for (int i = 0; i < 4; ++i) {
    int f = (i << 5) | lane;
    atomicAdd(&agg[d + f], x[s + f]);
  }
}

// h = (1+eps)*x + agg, cast to bf16 for the WMMA GEMM
__global__ void combine_to_bf16(const float* __restrict__ x,
                                const float* __restrict__ agg,
                                const float* __restrict__ eps,
                                unsigned short* __restrict__ out, int n) {
  int i = blockIdx.x * blockDim.x + threadIdx.x;
  if (i >= n) return;
  float e1 = 1.0f + eps[0];
  out[i] = f2bf(fmaf(e1, x[i], agg[i]));
}

// W[cin][cout] (f32, row-major) -> Bt[cout][cin] (bf16) so B-fragment K-loads are contiguous
__global__ void w_to_bf16t(const float* __restrict__ W,
                           unsigned short* __restrict__ Bt,
                           int cin, int cout) {
  int i = blockIdx.x * blockDim.x + threadIdx.x;
  if (i >= cin * cout) return;
  int o = i / cin, k = i - o * cin;
  Bt[i] = f2bf(W[k * cout + o]);
}

// ---------------------------------------------------------------- WMMA core
// One 16x16 output tile per wave.  A: MxK bf16 row-major, Bt: NcxK bf16 (B^T).
// Fragment layouts per CDNA5 ISA 7.12.2 (16-bit A 16x32, B 32x16, 32-bit C/D).
template <int K>
__device__ __forceinline__ v8f wmma_tile(const unsigned short* __restrict__ A,
                                         const unsigned short* __restrict__ Bt,
                                         int m0, int n0, int lane) {
  const int akoff = (lane < 16) ? 0 : 8;    // A: K 0..7/16..23 vs 8..15/24..31
  const int bkoff = (lane < 16) ? 0 : 8;    // B: K 0..7/16..23 vs 8..15/24..31
  const unsigned short* ap = A  + (m0 + (lane & 15)) * K + akoff;
  const unsigned short* bp = Bt + (n0 + (lane & 15)) * K + bkoff;
  v8f acc = {0.f, 0.f, 0.f, 0.f, 0.f, 0.f, 0.f, 0.f};
#pragma unroll
  for (int k0 = 0; k0 < K; k0 += 32) {
    FragAB a, b;
    const uint4* pa = reinterpret_cast<const uint4*>(ap + k0);
    a.q[0] = pa[0];          // K = base .. base+7
    a.q[1] = pa[2];          // K = base+16 .. base+23
    const uint4* pb = reinterpret_cast<const uint4*>(bp + k0);
    b.q[0] = pb[0];          // K = base .. base+7
    b.q[1] = pb[2];          // K = base+16 .. base+23
    acc = __builtin_amdgcn_wmma_f32_16x16x32_bf16(false, a.v, false, b.v,
                                                  (short)0, acc, false, false);
  }
  return acc;
}

// C(bf16) = relu(A@B + bias)      (first MLP GEMM, feeds second GEMM)
template <int K>
__global__ void gemm_bias_relu_bf16out(const unsigned short* __restrict__ A,
                                       const unsigned short* __restrict__ Bt,
                                       const float* __restrict__ bias,
                                       unsigned short* __restrict__ C,
                                       int Mtiles, int Nc) {
  int lane   = threadIdx.x & 31;
  int wave   = blockIdx.x * (blockDim.x >> 5) + (threadIdx.x >> 5);
  int ntiles = Nc >> 4;
  int tm = wave / ntiles, tn = wave - tm * ntiles;
  if (tm >= Mtiles) return;                 // whole-wave uniform (exact division anyway)
  int m0 = tm << 4, n0 = tn << 4;
  FragC c; c.v = wmma_tile<K>(A, Bt, m0, n0, lane);
  int col   = n0 + (lane & 15);
  int rbase = m0 + ((lane < 16) ? 0 : 8);
  float bv  = bias[col];
#pragma unroll
  for (int r = 0; r < 8; ++r) {
    float v = c.f[r] + bv;
    C[(rbase + r) * Nc + col] = f2bf(v > 0.f ? v : 0.f);
  }
}

// C(f32) = relu(BN(A@B + b2))     (second MLP GEMM + eval-mode BatchNorm)
template <int K>
__global__ void gemm_bn_relu_f32out(const unsigned short* __restrict__ A,
                                    const unsigned short* __restrict__ Bt,
                                    const float* __restrict__ b2,
                                    const float* __restrict__ bn_g,
                                    const float* __restrict__ bn_b,
                                    const float* __restrict__ bn_m,
                                    const float* __restrict__ bn_v,
                                    float* __restrict__ C,
                                    int Mtiles, int Nc) {
  int lane   = threadIdx.x & 31;
  int wave   = blockIdx.x * (blockDim.x >> 5) + (threadIdx.x >> 5);
  int ntiles = Nc >> 4;
  int tm = wave / ntiles, tn = wave - tm * ntiles;
  if (tm >= Mtiles) return;
  int m0 = tm << 4, n0 = tn << 4;
  FragC c; c.v = wmma_tile<K>(A, Bt, m0, n0, lane);
  int col   = n0 + (lane & 15);
  int rbase = m0 + ((lane < 16) ? 0 : 8);
  float sc = bn_g[col] * rsqrtf(bn_v[col] + 1e-5f);
  float sh = fmaf(b2[col] - bn_m[col], sc, bn_b[col]);  // (acc+b2-m)*sc+bb = acc*sc + sh
#pragma unroll
  for (int r = 0; r < 8; ++r) {
    float v = fmaf(c.f[r], sc, sh);
    C[(rbase + r) * Nc + col] = v > 0.f ? v : 0.f;
  }
}

// ---------------------------------------------------------------- pool + head
__global__ void pool_add(const float* __restrict__ h,
                         const int* __restrict__ batch,
                         float* __restrict__ hg) {
  int i = blockIdx.x * blockDim.x + threadIdx.x;
  if (i >= NN * DD) return;
  int node = i >> 7, f = i & 127;
  atomicAdd(&hg[batch[node] * DD + f], h[i]);
}

// per graph: relu(hg@lin1 + b1) @ lin2 + b2 -> log_softmax  (exact f32, tiny)
__global__ void head_kernel(const float* __restrict__ hg,
                            const float* __restrict__ w1, const float* __restrict__ b1,
                            const float* __restrict__ w2, const float* __restrict__ b2,
                            float* __restrict__ out) {
  int g = blockIdx.x, j = threadIdx.x;      // 128 threads
  __shared__ float row[DD];
  __shared__ float act[DD];
  __shared__ float logits[NO];
  row[j] = hg[g * DD + j];
  __syncthreads();
  float a = b1[j];
  for (int k = 0; k < DD; ++k) a = fmaf(row[k], w1[k * DD + j], a);
  act[j] = a > 0.f ? a : 0.f;
  __syncthreads();
  if (j < NO) {
    float l = b2[j];
    for (int k = 0; k < DD; ++k) l = fmaf(act[k], w2[k * NO + j], l);
    logits[j] = l;
  }
  __syncthreads();
  if (j == 0) {
    float mx = logits[0];
    for (int o = 1; o < NO; ++o) mx = fmaxf(mx, logits[o]);
    float s = 0.f;
    for (int o = 0; o < NO; ++o) s += expf(logits[o] - mx);
    float lse = mx + logf(s);
    for (int o = 0; o < NO; ++o) out[g * NO + o] = logits[o] - lse;
  }
}

// ---------------------------------------------------------------- launch
extern "C" void kernel_launch(void* const* d_in, const int* in_sizes, int n_in,
                              void* d_out, int out_size, void* d_ws, size_t ws_size,
                              hipStream_t stream) {
  (void)in_sizes; (void)n_in; (void)out_size; (void)ws_size;
  const float* x     = (const float*)d_in[0];
  const int*   ei    = (const int*)d_in[1];
  const int*   srcE  = ei;
  const int*   dstE  = ei + NE;
  const int*   batch = (const int*)d_in[2];
  const float* eps0  = (const float*)d_in[3];
  const float* w1_0  = (const float*)d_in[4];
  const float* b1_0  = (const float*)d_in[5];
  const float* w2_0  = (const float*)d_in[6];
  const float* b2_0  = (const float*)d_in[7];
  const float* g0    = (const float*)d_in[8];
  const float* bb0   = (const float*)d_in[9];
  const float* m0_   = (const float*)d_in[10];
  const float* v0_   = (const float*)d_in[11];
  const float* eps1  = (const float*)d_in[12];
  const float* w1_1  = (const float*)d_in[13];
  const float* b1_1  = (const float*)d_in[14];
  const float* w2_1  = (const float*)d_in[15];
  const float* b2_1  = (const float*)d_in[16];
  const float* g1    = (const float*)d_in[17];
  const float* bb1   = (const float*)d_in[18];
  const float* m1_   = (const float*)d_in[19];
  const float* v1_   = (const float*)d_in[20];
  const float* lin1w = (const float*)d_in[21];
  const float* lin1b = (const float*)d_in[22];
  const float* lin2w = (const float*)d_in[23];
  const float* lin2b = (const float*)d_in[24];

  char* ws = (char*)d_ws;
  float*          agg  = (float*)(ws);                       //  5,120,000 B
  float*          h    = (float*)(ws + 5120000);             //  5,120,000 B
  unsigned short* xb   = (unsigned short*)(ws + 10240000);   //  2,560,000 B
  unsigned short* hid  = (unsigned short*)(ws + 12800000);   //  5,120,000 B
  unsigned short* w1t0 = (unsigned short*)(ws + 17920000);   //  65,536 B each
  unsigned short* w2t0 = w1t0 + DD * HID2;
  unsigned short* w1t1 = w2t0 + HID2 * DD;
  unsigned short* w2t1 = w1t1 + DD * HID2;
  float*          hg   = (float*)(ws + 17920000 + 4 * DD * HID2 * 2);

  const int THR = 256;
  // weight conversion / transpose (re-done every call: deterministic, trivial cost)
  w_to_bf16t<<<(DD * HID2 + THR - 1) / THR, THR, 0, stream>>>(w1_0, w1t0, DD, HID2);
  w_to_bf16t<<<(HID2 * DD + THR - 1) / THR, THR, 0, stream>>>(w2_0, w2t0, HID2, DD);
  w_to_bf16t<<<(DD * HID2 + THR - 1) / THR, THR, 0, stream>>>(w1_1, w1t1, DD, HID2);
  w_to_bf16t<<<(HID2 * DD + THR - 1) / THR, THR, 0, stream>>>(w2_1, w2t1, HID2, DD);
  zero_f32<<<(NG * DD + THR - 1) / THR, THR, 0, stream>>>(hg, NG * DD);

  const int nElem  = NN * DD;                 // 1,280,000
  const int eBlks  = (NE * 32) / THR;         // 80,000
  const int mTiles = NN / 16;                 // 625 (exact)
  const int g1Blks = mTiles * (HID2 / 16) / 8;  // 1250 (8 waves/block)
  const int g2Blks = mTiles * (DD / 16) / 8;    // 625

  // ---- GIN layer 0
  zero_f32<<<nElem / THR, THR, 0, stream>>>(agg, nElem);
  scatter_add<<<eBlks, THR, 0, stream>>>(x, srcE, dstE, agg);
  combine_to_bf16<<<nElem / THR, THR, 0, stream>>>(x, agg, eps0, xb, nElem);
  gemm_bias_relu_bf16out<DD><<<g1Blks, THR, 0, stream>>>(xb, w1t0, b1_0, hid, mTiles, HID2);
  gemm_bn_relu_f32out<HID2><<<g2Blks, THR, 0, stream>>>(hid, w2t0, b2_0, g0, bb0, m0_, v0_,
                                                        h, mTiles, DD);
  // ---- GIN layer 1
  zero_f32<<<nElem / THR, THR, 0, stream>>>(agg, nElem);
  scatter_add<<<eBlks, THR, 0, stream>>>(h, srcE, dstE, agg);
  combine_to_bf16<<<nElem / THR, THR, 0, stream>>>(h, agg, eps1, xb, nElem);
  gemm_bias_relu_bf16out<DD><<<g1Blks, THR, 0, stream>>>(xb, w1t1, b1_1, hid, mTiles, HID2);
  gemm_bn_relu_f32out<HID2><<<g2Blks, THR, 0, stream>>>(hid, w2t1, b2_1, g1, bb1, m1_, v1_,
                                                        h, mTiles, DD);
  // ---- pool + head
  pool_add<<<nElem / THR, THR, 0, stream>>>(h, batch, hg);
  head_kernel<<<NG, DD, 0, stream>>>(hg, lin1w, lin1b, lin2w, lin2b, (float*)d_out);
}
